// cosine_similarity_87101936763000
// MI455X (gfx1250) — matrix operand order repaired and checked by reading, not compiled, not run on hardware
//
#include <hip/hip_runtime.h>
#include <hip/hip_bf16.h>

typedef float  v8f  __attribute__((ext_vector_type(8)));
typedef float  f4   __attribute__((ext_vector_type(4)));
typedef __bf16 v8bf __attribute__((ext_vector_type(8)));
typedef __bf16 v16bf __attribute__((ext_vector_type(16)));
typedef unsigned int v2u __attribute__((ext_vector_type(2)));

#define SEQ   1024
#define NB    32
#define DIM   128
#define ROWS_PER_BLOCK 16
#define ROW_TILES (SEQ / ROWS_PER_BLOCK)     // 64
#define WAVES 4
#define THREADS (WAVES * 32)
#define TILES_PER_WAVE (SEQ / (WAVES * 16))  // 16
#define SPAD  1028                           // f32 scores row stride (pad: half-waves on disjoint banks)
#define BPAD  136                            // bf16 panel row stride (272B -> 4-bank row skew)

// LDS byte layout
#define SCORES_BYTES (ROWS_PER_BLOCK * SPAD * 4)                 // 65792
#define PANEL_ELEMS  (16 * BPAD)                                 // ushorts per 16x128 panel
#define SMEM_BYTES   (SCORES_BYTES + (2 * PANEL_ELEMS + 2 * WAVES * PANEL_ELEMS) * 2 + 16)

// split one float4 into packed-bf16 hi (2 dwords) and lo (2 dwords); hi by truncation
// (lo = v - hi is then exact in f32), lo also truncated to bf16.
__device__ __forceinline__ void split4(const f4 v, v2u& hi, v2u& lo) {
    unsigned u0 = __float_as_uint(v[0]);
    unsigned u1 = __float_as_uint(v[1]);
    unsigned u2 = __float_as_uint(v[2]);
    unsigned u3 = __float_as_uint(v[3]);
    hi[0] = (u0 >> 16) | (u1 & 0xffff0000u);
    hi[1] = (u2 >> 16) | (u3 & 0xffff0000u);
    float l0 = v[0] - __uint_as_float(u0 & 0xffff0000u);
    float l1 = v[1] - __uint_as_float(u1 & 0xffff0000u);
    float l2 = v[2] - __uint_as_float(u2 & 0xffff0000u);
    float l3 = v[3] - __uint_as_float(u3 & 0xffff0000u);
    unsigned w0 = __float_as_uint(l0), w1 = __float_as_uint(l1);
    unsigned w2 = __float_as_uint(l2), w3 = __float_as_uint(l3);
    lo[0] = (w0 >> 16) | (w1 & 0xffff0000u);
    lo[1] = (w2 >> 16) | (w3 & 0xffff0000u);
}

// assemble a v16bf fragment from two 8-element runs at p and p+gap
__device__ __forceinline__ v16bf ldfrag(const unsigned short* p, int gap) {
    v8bf a = *(const v8bf*)p;
    v8bf b = *(const v8bf*)(p + gap);
    return __builtin_shufflevector(a, b, 0, 1, 2, 3, 4, 5, 6, 7,
                                         8, 9, 10, 11, 12, 13, 14, 15);
}

__global__ void __launch_bounds__(THREADS)
attn_pool_kernel(const float* __restrict__ X, const float* __restrict__ Y,
                 float* __restrict__ partial) {
    extern __shared__ char smem[];
    float* scores          = (float*)smem;                         // 16 x 1028 f32
    unsigned short* Yh     = (unsigned short*)(smem + SCORES_BYTES);
    unsigned short* Yl     = Yh + PANEL_ELEMS;
    unsigned short* Xh     = Yl + PANEL_ELEMS;                     // WAVES panels
    unsigned short* Xl     = Xh + WAVES * PANEL_ELEMS;
    float* red             = (float*)(Xl + WAVES * PANEL_ELEMS);

    const int tid  = threadIdx.x;
    const int lane = tid & 31;
    const int wave = tid >> 5;
    const int half = lane >> 4;      // half-wave id
    const int lr   = lane & 15;      // row/col within tile

    const int n        = blockIdx.y;
    const int rt       = blockIdx.x;
    const int row_base = rt * ROWS_PER_BLOCK;

    // ---- stage Y stripe (16 x 128), split f32 -> bf16 hi/lo panels ----
    for (int i = tid; i < ROWS_PER_BLOCK * 32; i += THREADS) {
        int row = i >> 5;
        int c4  = i & 31;
        const f4 yv = *(const f4*)(Y + (((size_t)(row_base + row) * NB + n) * DIM) + c4 * 4);
        v2u hi, lo;
        split4(yv, hi, lo);
        *(v2u*)((unsigned int*)Yh + row * (BPAD / 2) + c4 * 2) = hi;
        *(v2u*)((unsigned int*)Yl + row * (BPAD / 2) + c4 * 2) = lo;
    }
    __syncthreads();

    // ---- preload A fragments: 4 K-chunks, hi and lo ----
    // 16-bit A 16x32 layout: lane<16 holds K {8h..8h+7} U {16+8h..16+8h+7}, h=half
    v16bf ah[4], al[4];
#pragma unroll
    for (int c = 0; c < 4; ++c) {
        const int base = lr * BPAD + 32 * c + 8 * half;
        ah[c] = ldfrag(Yh + base, 16);
        al[c] = ldfrag(Yl + base, 16);
    }

    unsigned short* Xwh = Xh + wave * PANEL_ELEMS;
    unsigned short* Xwl = Xl + wave * PANEL_ELEMS;
    unsigned int*   XwhU = (unsigned int*)Xwh;
    unsigned int*   XwlU = (unsigned int*)Xwl;

    // ---- each wave: 16 column tiles ----
    for (int t = 0; t < TILES_PER_WAVE; ++t) {
        const int col_base = (wave * TILES_PER_WAVE + t) * 16;

        // stage X panel (16 x 128) with on-the-fly hi/lo split, coalesced
        for (int r = 0; r < 16; ++r) {
            const f4 xv = *(const f4*)(X + (((size_t)(col_base + r) * NB + n) * DIM) + lane * 4);
            v2u hi, lo;
            split4(xv, hi, lo);
            *(v2u*)(XwhU + r * (BPAD / 2) + lane * 2) = hi;
            *(v2u*)(XwlU + r * (BPAD / 2) + lane * 2) = lo;
        }

        v8f acc0 = {0.f, 0.f, 0.f, 0.f, 0.f, 0.f, 0.f, 0.f};  // hi*hi
        v8f acc1 = {0.f, 0.f, 0.f, 0.f, 0.f, 0.f, 0.f, 0.f};  // cross terms
#pragma unroll
        for (int c = 0; c < 4; ++c) {
            // 16-bit B 32x16 layout: lane half h holds K {8h..8h+7} U {16+8h..16+8h+7} (matches A)
            const int bbase = lr * BPAD + 32 * c + 8 * half;
            v16bf bh = ldfrag(Xwh + bbase, 16);
            v16bf bl = ldfrag(Xwl + bbase, 16);
            acc0 = __builtin_amdgcn_wmma_f32_16x16x32_bf16(
                false, ah[c], false, bh, (short)0, acc0, false, false);
            acc1 = __builtin_amdgcn_wmma_f32_16x16x32_bf16(
                false, al[c], false, bh, (short)0, acc1, false, false);
            acc1 = __builtin_amdgcn_wmma_f32_16x16x32_bf16(
                false, ah[c], false, bl, (short)0, acc1, false, false);
        }
        v8f acc = acc0 + acc1;

        // C layout: lanes 0-15 -> M=v, N=lane ; lanes 16-31 -> M=8+v, N=lane-16
#pragma unroll
        for (int v = 0; v < 8; ++v)
            scores[(half * 8 + v) * SPAD + col_base + lr] = acc[v];
    }
    __syncthreads();

    // ---- per-row: rowmax -> exp/sum -> window-of-3 max (wave-local) ----
    float acc_rows = 0.f;
    for (int rr = 0; rr < 4; ++rr) {
        const int r = wave * 4 + rr;
        float* srow = scores + r * SPAD;

        float m = -INFINITY;
        for (int j = lane; j < SEQ; j += 32) m = fmaxf(m, srow[j]);
#pragma unroll
        for (int off = 16; off >= 1; off >>= 1) m = fmaxf(m, __shfl_xor(m, off, 32));

        float lsum = 0.f;
        for (int j = lane; j < SEQ; j += 32) {
            float e = __expf(srow[j] - m);
            srow[j] = e;
            lsum += e;
        }
#pragma unroll
        for (int off = 16; off >= 1; off >>= 1) lsum += __shfl_xor(lsum, off, 32);

        float lw = -INFINITY;
        for (int j = lane; j < SEQ - 2; j += 32) {
            float w = srow[j] + srow[j + 1] + srow[j + 2];
            lw = fmaxf(lw, w);
        }
#pragma unroll
        for (int off = 16; off >= 1; off >>= 1) lw = fmaxf(lw, __shfl_xor(lw, off, 32));

        acc_rows += lw / (3.0f * lsum);
    }

    if (lane == 0) red[wave] = acc_rows;
    __syncthreads();
    if (tid == 0)
        partial[n * ROW_TILES + rt] = red[0] + red[1] + red[2] + red[3];
}

__global__ void finalize_kernel(const float* __restrict__ partial, float* __restrict__ out) {
    __shared__ float s[ROW_TILES];
    const int n = blockIdx.x;
    const int t = threadIdx.x;
    s[t] = partial[n * ROW_TILES + t];
    __syncthreads();
    for (int off = ROW_TILES / 2; off >= 1; off >>= 1) {
        if (t < off) s[t] += s[t + off];
        __syncthreads();
    }
    if (t == 0) out[n] = s[0] / (float)SEQ;
}

extern "C" void kernel_launch(void* const* d_in, const int* in_sizes, int n_in,
                              void* d_out, int out_size, void* d_ws, size_t ws_size,
                              hipStream_t stream) {
    const float* x = (const float*)d_in[0];  // (1024, 32, 128)
    const float* y = (const float*)d_in[1];  // (1024, 32, 128)
    float* out     = (float*)d_out;          // (32,)
    float* partial = (float*)d_ws;           // 32*64 floats

    dim3 grid(ROW_TILES, NB);
    attn_pool_kernel<<<grid, THREADS, SMEM_BYTES, stream>>>(x, y, partial);
    finalize_kernel<<<NB, ROW_TILES, 0, stream>>>(partial, out);
}
